// LocalfeatureIntegrationTransformer_7696581394668
// MI455X (gfx1250) — matrix operand order repaired and checked by reading, not compiled, not run on hardware
//
#include <hip/hip_runtime.h>

typedef __bf16 bf16_t;
typedef __bf16 v4bf  __attribute__((ext_vector_type(4)));
typedef __bf16 v8bf  __attribute__((ext_vector_type(8)));
typedef __bf16 v16bf __attribute__((ext_vector_type(16)));
typedef float  v8f   __attribute__((ext_vector_type(8)));
typedef __attribute__((address_space(3))) char lds_char;

// Flip to 0 to fall back to the sync global->VGPR->LDS staging.
#define USE_ASYNC_LDS 1

namespace {
constexpr int kIters = 6;
constexpr int kNS   = 256;    // N_SLOTS
constexpr int kD    = 1024;   // DIM == INPUT_DIM
constexpr int kB    = 8;
constexpr int kL    = 4096;   // H*W
constexpr float kScale = 0.03125f;  // DIM^-0.5

constexpr int BM = 128, BN = 64, BK = 64;
constexpr int LDA_S = BK + 16;   // 80 bf16 (160B rows, 16B-aligned chunks)
constexpr int LDB_S = BN + 16;   // 80 bf16
}

// 16B global -> LDS copy; async (gfx1250 GLOBAL_LOAD_ASYNC_TO_LDS_B128) or sync.
__device__ __forceinline__ void copy16_to_lds(void* lds, const void* g) {
#if USE_ASYNC_LDS
  unsigned la = (unsigned)(unsigned long long)(lds_char*)lds;
  asm volatile("global_load_async_to_lds_b128 %0, %1, off"
               :: "v"(la), "v"(g) : "memory");
#else
  *(uint4*)lds = *(const uint4*)g;
#endif
}

__device__ __forceinline__ void wait_async_group_pending() {
#if USE_ASYNC_LDS
  asm volatile("s_wait_asynccnt 6" ::: "memory");   // next stage (6 ops) in flight
#endif
}
__device__ __forceinline__ void wait_async_all() {
#if USE_ASYNC_LDS
  asm volatile("s_wait_asynccnt 0" ::: "memory");
#endif
}

// ---------------------------------------------------------------------------
// Generic bf16 WMMA GEMM: acc_f32 = A[M,K] * B[K,N]; epilogue by `mode`.
//  mode 0: Cf = acc*alpha                       (f32, row-major ldc)
//  mode 1: Cb = bf16(acc*alpha)                 (bf16, row-major ldcb)
//  mode 2: per-256-row batch transposed store:  qT[b][n][m%256] = bf16(acc*alpha)
//  mode 3: Cb = bf16(relu(acc*alpha + bias[n]))
//  mode 4: Cf += acc*alpha + bias[n]            (residual accumulate)
// Double-buffered LDS, async tile staging, 8 WMMAs per stage.
// ---------------------------------------------------------------------------
__global__ __launch_bounds__(256)
void gemm_bf16_wmma(const bf16_t* __restrict__ A, long lda, long strideA,
                    const bf16_t* __restrict__ Bm, long ldb, long strideB,
                    float* Cf, long ldc, long strideC,
                    bf16_t* Cb, long ldcb, long strideCb,
                    const float* __restrict__ bias, float alpha,
                    int M, int N, int K, int mode) {
  __shared__ bf16_t sA[2][BM][LDA_S];
  __shared__ bf16_t sB[2][BK][LDB_S];

  const int tid  = threadIdx.x;
  const int lane = tid & 31;
  const int wave = tid >> 5;
  const int wm   = wave & 3;   // 4 waves along M -> 32 rows each
  const int wn   = wave >> 2;  // 2 waves along N -> 32 cols each
  const int m0   = blockIdx.y * BM;
  const int n0   = blockIdx.x * BN;
  const int bz   = blockIdx.z;

  A  += (long)bz * strideA;
  Bm += (long)bz * strideB;

  v8f acc[2][2];
#pragma unroll
  for (int mt = 0; mt < 2; ++mt)
#pragma unroll
    for (int nt = 0; nt < 2; ++nt)
      acc[mt][nt] = v8f{0.f, 0.f, 0.f, 0.f, 0.f, 0.f, 0.f, 0.f};

  const int lr = lane & 15;   // row within 16
  const int lh = lane >> 4;   // K-half selector for A fragments

  // Stage one BK tile (A: 128x64, B: 64x64 bf16) = 6 async b128 ops per wave.
  auto stage = [&](int kt, int buf) {
    const long k0 = (long)kt * BK;
#pragma unroll
    for (int i = 0; i < 4; ++i) {          // A: 1024 x 16B chunks / 256 thr
      int ch = tid + i * 256;
      int r = ch >> 3, c = (ch & 7) * 8;
      copy16_to_lds(&sA[buf][r][c], A + (long)(m0 + r) * lda + k0 + c);
    }
#pragma unroll
    for (int i = 0; i < 2; ++i) {          // B: 512 x 16B chunks / 256 thr
      int ch = tid + i * 256;
      int r = ch >> 3, c = (ch & 7) * 8;
      copy16_to_lds(&sB[buf][r][c], Bm + (k0 + r) * ldb + n0 + c);
    }
  };

  const int nk = K / BK;
  stage(0, 0);
  for (int kt = 0; kt < nk; ++kt) {
    const int cur = kt & 1;
    if (kt + 1 < nk) {
      stage(kt + 1, cur ^ 1);
      wait_async_group_pending();   // tile kt complete; kt+1 still in flight
    } else {
      wait_async_all();
    }
    __syncthreads();

#pragma unroll
    for (int ks = 0; ks < 2; ++ks) {       // two 16x16x32 K-steps per tile
      v16bf af[2], bfrag[2];
#pragma unroll
      for (int mt = 0; mt < 2; ++mt) {
        int row = wm * 32 + mt * 16 + lr;
        v8bf lo = *(const v8bf*)(&sA[cur][row][ks * 32 + lh * 8]);
        v8bf hi = *(const v8bf*)(&sA[cur][row][ks * 32 + 16 + lh * 8]);
        af[mt] = __builtin_shufflevector(lo, hi, 0, 1, 2, 3, 4, 5, 6, 7,
                                         8, 9, 10, 11, 12, 13, 14, 15);
      }
#pragma unroll
      for (int nt = 0; nt < 2; ++nt) {
        int col = wn * 32 + nt * 16;
        v8bf lo, hi;
#pragma unroll
        for (int j = 0; j < 8; ++j) {
          lo[j] = sB[cur][ks * 32 + lh * 8 + j][col + lr];
          hi[j] = sB[cur][ks * 32 + 16 + lh * 8 + j][col + lr];
        }
        bfrag[nt] = __builtin_shufflevector(lo, hi, 0, 1, 2, 3, 4, 5, 6, 7,
                                            8, 9, 10, 11, 12, 13, 14, 15);
      }
#pragma unroll
      for (int mt = 0; mt < 2; ++mt)
#pragma unroll
        for (int nt = 0; nt < 2; ++nt)
          acc[mt][nt] = __builtin_amdgcn_wmma_f32_16x16x32_bf16(
              false, af[mt], false, bfrag[nt], (short)0, acc[mt][nt], false, false);
    }
    __syncthreads();
  }

  // ---- epilogue ----
#pragma unroll
  for (int mt = 0; mt < 2; ++mt) {
#pragma unroll
    for (int nt = 0; nt < 2; ++nt) {
      const int mbase = m0 + wm * 32 + mt * 16 + (lane >> 4) * 8;
      const int n     = n0 + wn * 32 + nt * 16 + (lane & 15);
      v8f v = acc[mt][nt];
      if (mode == 0) {
        float* p = Cf + (long)bz * strideC;
#pragma unroll
        for (int i = 0; i < 8; ++i)
          p[(long)(mbase + i) * ldc + n] = v[i] * alpha;
      } else if (mode == 1) {
        bf16_t* p = Cb + (long)bz * strideCb;
#pragma unroll
        for (int i = 0; i < 8; ++i)
          p[(long)(mbase + i) * ldcb + n] = (bf16_t)(v[i] * alpha);
      } else if (mode == 2) {
        v8bf pk;
#pragma unroll
        for (int i = 0; i < 8; ++i) pk[i] = (bf16_t)(v[i] * alpha);
        const int bb = mbase >> 8;          // batch within fused M (256 rows each)
        const int slot = mbase & 255;
        bf16_t* p = Cb + (long)bb * strideCb + (long)n * ldcb + slot;
        *(uint4*)p = *(uint4*)&pk;          // 8 consecutive slots, 16B store
      } else if (mode == 3) {
        const float bv = bias[n];
#pragma unroll
        for (int i = 0; i < 8; ++i) {
          float t = v[i] * alpha + bv;
          Cb[(long)(mbase + i) * ldcb + n] = (bf16_t)fmaxf(t, 0.f);
        }
      } else {  // mode 4
        const float bv = bias ? bias[n] : 0.f;
        float* p = Cf + (long)bz * strideC;
#pragma unroll
        for (int i = 0; i < 8; ++i) {
          long idx = (long)(mbase + i) * ldc + n;
          p[idx] = p[idx] + v[i] * alpha + bv;
        }
      }
    }
  }
}

// ---------------------------------------------------------------------------
// Input LN over C: x[B,C,L] -> xn bf16 [B*L, C]. One thread per (b,l) column.
// ---------------------------------------------------------------------------
__global__ __launch_bounds__(256)
void ln_input_kernel(const float* __restrict__ x, const float* __restrict__ g,
                     const float* __restrict__ bt, bf16_t* __restrict__ xn) {
  const int l = blockIdx.x * 256 + threadIdx.x;
  const int b = blockIdx.y;
  const float* xb = x + (long)b * kD * kL + l;
  float s = 0.f, ss = 0.f;
#pragma unroll 4
  for (int c = 0; c < kD; ++c) {
    float v = xb[(long)c * kL];
    s += v; ss += v * v;
  }
  const float mean = s * (1.f / kD);
  const float var  = ss * (1.f / kD) - mean * mean;
  const float inv  = rsqrtf(var + 1e-5f);
  bf16_t* o = xn + ((long)b * kL + l) * kD;
#pragma unroll 4
  for (int c = 0; c < kD; ++c) {
    float v = (xb[(long)c * kL] - mean) * inv * g[c] + bt[c];
    o[c] = (bf16_t)v;
  }
}

// ---------------------------------------------------------------------------
// Row LN: in f32 [rows,1024] -> out bf16 [rows,1024]. One block per row.
// ---------------------------------------------------------------------------
__global__ __launch_bounds__(256)
void ln_rows_kernel(const float* __restrict__ in, const float* __restrict__ g,
                    const float* __restrict__ bt, bf16_t* __restrict__ out) {
  const long row = blockIdx.x;
  const int tid = threadIdx.x;
  const float4 v = ((const float4*)(in + row * kD))[tid];
  __shared__ float r1[256], r2[256];
  r1[tid] = v.x + v.y + v.z + v.w;
  r2[tid] = v.x * v.x + v.y * v.y + v.z * v.z + v.w * v.w;
  __syncthreads();
  for (int off = 128; off > 0; off >>= 1) {
    if (tid < off) { r1[tid] += r1[tid + off]; r2[tid] += r2[tid + off]; }
    __syncthreads();
  }
  const float mean = r1[0] * (1.f / kD);
  const float var  = r2[0] * (1.f / kD) - mean * mean;
  const float inv  = rsqrtf(var + 1e-5f);
  const float4 gg = ((const float4*)g)[tid];
  const float4 bb = ((const float4*)bt)[tid];
  v4bf o;
  o[0] = (bf16_t)((v.x - mean) * inv * gg.x + bb.x);
  o[1] = (bf16_t)((v.y - mean) * inv * gg.y + bb.y);
  o[2] = (bf16_t)((v.z - mean) * inv * gg.z + bb.z);
  o[3] = (bf16_t)((v.w - mean) * inv * gg.w + bb.w);
  *(v4bf*)(out + row * kD + tid * 4) = o;
}

// ---------------------------------------------------------------------------
// Softmax over N=256 per row, in-place, +1e-8. One wave32 per row.
// ---------------------------------------------------------------------------
__global__ __launch_bounds__(256)
void softmax_rows_kernel(float* __restrict__ logits) {
  const long row = (long)blockIdx.x * 8 + (threadIdx.x >> 5);
  const int lane = threadIdx.x & 31;
  float* p = logits + row * kNS;
  float4 a = ((float4*)p)[lane];
  float4 b = ((float4*)p)[lane + 32];
  float mx = fmaxf(fmaxf(fmaxf(a.x, a.y), fmaxf(a.z, a.w)),
                   fmaxf(fmaxf(b.x, b.y), fmaxf(b.z, b.w)));
  for (int off = 16; off > 0; off >>= 1) mx = fmaxf(mx, __shfl_xor(mx, off, 32));
  a.x = __expf(a.x - mx); a.y = __expf(a.y - mx);
  a.z = __expf(a.z - mx); a.w = __expf(a.w - mx);
  b.x = __expf(b.x - mx); b.y = __expf(b.y - mx);
  b.z = __expf(b.z - mx); b.w = __expf(b.w - mx);
  float s = a.x + a.y + a.z + a.w + b.x + b.y + b.z + b.w;
  for (int off = 16; off > 0; off >>= 1) s += __shfl_xor(s, off, 32);
  const float r = 1.f / s;
  a.x = a.x * r + 1e-8f; a.y = a.y * r + 1e-8f;
  a.z = a.z * r + 1e-8f; a.w = a.w * r + 1e-8f;
  b.x = b.x * r + 1e-8f; b.y = b.y * r + 1e-8f;
  b.z = b.z * r + 1e-8f; b.w = b.w * r + 1e-8f;
  ((float4*)p)[lane] = a;
  ((float4*)p)[lane + 32] = b;
}

__global__ __launch_bounds__(256)
void zero_kernel(float* __restrict__ p, int n) {
  int i = blockIdx.x * 256 + threadIdx.x;
  if (i < n) p[i] = 0.f;
}

// Column sums over L (renorm denominator). grid (L/128, B); thread = column n.
__global__ __launch_bounds__(256)
void colsum_kernel(const float* __restrict__ attn, float* __restrict__ colsum) {
  const int n = threadIdx.x;
  const int b = blockIdx.y;
  const int l0 = blockIdx.x * 128;
  const float* p = attn + ((long)b * kL + l0) * kNS + n;
  float s = 0.f;
#pragma unroll 4
  for (int i = 0; i < 128; ++i) s += p[(long)i * kNS];
  atomicAdd(&colsum[b * kNS + n], s);
}

// attn[b,l,n]/colsum[b,n] -> attnT bf16 [b,n,l] and f32 (the attn output).
__global__ __launch_bounds__(256)
void renorm_transpose_kernel(const float* __restrict__ attn,
                             const float* __restrict__ colsum,
                             bf16_t* __restrict__ attnT,
                             float* __restrict__ attnT_f32) {
  __shared__ float tile[64][65];
  const int b = blockIdx.z, l0 = blockIdx.x * 64, n0 = blockIdx.y * 64;
  const int tid = threadIdx.x;
#pragma unroll
  for (int i = 0; i < 16; ++i) {
    int e = tid + i * 256;
    int r = e >> 6, c = e & 63;
    tile[r][c] = attn[((long)b * kL + l0 + r) * kNS + n0 + c] /
                 colsum[b * kNS + n0 + c];
  }
  __syncthreads();
#pragma unroll
  for (int i = 0; i < 16; ++i) {
    int e = tid + i * 256;
    int nr = e >> 6, lc = e & 63;
    float v = tile[lc][nr];
    long o = ((long)b * kNS + n0 + nr) * kL + l0 + lc;
    attnT[o] = (bf16_t)v;
    attnT_f32[o] = v;
  }
}

// templates [B,N,D] f32 -> out [B,D,N] f32 (64x64 LDS tiles)
__global__ __launch_bounds__(256)
void transpose_templates_kernel(const float* __restrict__ tmpl,
                                float* __restrict__ out) {
  __shared__ float tile[64][65];
  const int b = blockIdx.z, d0 = blockIdx.x * 64, n0 = blockIdx.y * 64;
  const int tid = threadIdx.x;
#pragma unroll
  for (int i = 0; i < 16; ++i) {
    int e = tid + i * 256;
    int r = e >> 6, c = e & 63;
    tile[r][c] = tmpl[((long)b * kNS + n0 + r) * kD + d0 + c];
  }
  __syncthreads();
#pragma unroll
  for (int i = 0; i < 16; ++i) {
    int e = tid + i * 256;
    int rd = e >> 6, cn = e & 63;
    out[((long)b * kD + d0 + rd) * kNS + n0 + cn] = tile[cn][rd];
  }
}

// f32 -> bf16 with destination row stride (weight conversion / packing)
__global__ __launch_bounds__(256)
void cvt2d_kernel(const float* __restrict__ src, bf16_t* __restrict__ dst,
                  int rows, int cols, int ldd) {
  long idx = (long)blockIdx.x * 256 + threadIdx.x;
  if (idx < (long)rows * cols) {
    int r = (int)(idx / cols), c = (int)(idx % cols);
    dst[(long)r * ldd + c] = (bf16_t)src[idx];
  }
}

__global__ __launch_bounds__(256)
void bcast_templates_kernel(const float* __restrict__ init,
                            float* __restrict__ tmpl) {
  long idx = (long)blockIdx.x * 256 + threadIdx.x;
  tmpl[idx] = init[idx & (kNS * kD - 1)];
}

// ---------------------------------------------------------------------------
extern "C" void kernel_launch(void* const* d_in, const int* in_sizes, int n_in,
                              void* d_out, int out_size, void* d_ws, size_t ws_size,
                              hipStream_t stream) {
  const float* x       = (const float*)d_in[0];
  const float* t_init  = (const float*)d_in[1];
  const float* Wq      = (const float*)d_in[2];
  const float* Wk      = (const float*)d_in[3];
  const float* Wv      = (const float*)d_in[4];
  const float* ln_in_g = (const float*)d_in[5];
  const float* ln_in_b = (const float*)d_in[6];
  const float* ln_t_g  = (const float*)d_in[7];
  const float* ln_t_b  = (const float*)d_in[8];
  const float* ln_m_g  = (const float*)d_in[9];
  const float* ln_m_b  = (const float*)d_in[10];
  const float* W1      = (const float*)d_in[11];
  const float* b1      = (const float*)d_in[12];
  const float* W2      = (const float*)d_in[13];
  const float* b2      = (const float*)d_in[14];
  float* out = (float*)d_out;

  // ---- workspace layout (~208 MB). xn region aliased per-iteration buffers.
  char* ws = (char*)d_ws;
  bf16_t* kv     = (bf16_t*)ws;                          // [32768, 2048] bf16 (k|v)
  char*   regA   = ws + 134217728;
  bf16_t* xn     = (bf16_t*)regA;                        // [32768, 1024] bf16
  float*  logits = (float*)regA;                         // [8, 4096, 256] f32 (alias)
  bf16_t* attnT  = (bf16_t*)(regA + 33554432);           // [8, 256, 4096] bf16
  bf16_t* qT     = (bf16_t*)(regA + 50331648);           // [8, 1024, 256] bf16
  bf16_t* tbf    = (bf16_t*)(regA + 54525952);           // [2048, 1024] bf16
  bf16_t* hbf    = (bf16_t*)(regA + 58720256);           // [2048, 512] bf16
  float*  colsum = (float*)(regA + 60817408);            // [8, 256] f32
  char*   wreg   = ws + 201326592;
  bf16_t* Wkv_b  = (bf16_t*)wreg;                        // [1024, 2048] bf16
  bf16_t* Wq_b   = (bf16_t*)(wreg + 4194304);            // [1024, 1024]
  bf16_t* W1_b   = (bf16_t*)(wreg + 6291456);            // [1024, 512]
  bf16_t* W2_b   = (bf16_t*)(wreg + 7340032);            // [512, 1024]
  float*  tmpl   = (float*)(wreg + 8388608);             // [8, 256, 1024] f32
  float*  out_attn = out + (long)kB * kD * kNS;          // [8, 256, 4096] f32

  // ---- weight conversion / packing
  cvt2d_kernel<<<4096, 256, 0, stream>>>(Wk, Wkv_b, 1024, 1024, 2048);
  cvt2d_kernel<<<4096, 256, 0, stream>>>(Wv, Wkv_b + 1024, 1024, 1024, 2048);
  cvt2d_kernel<<<4096, 256, 0, stream>>>(Wq, Wq_b, 1024, 1024, 1024);
  cvt2d_kernel<<<2048, 256, 0, stream>>>(W1, W1_b, 1024, 512, 512);
  cvt2d_kernel<<<2048, 256, 0, stream>>>(W2, W2_b, 512, 1024, 1024);

  // ---- input LN + templates init
  ln_input_kernel<<<dim3(kL / 256, kB), 256, 0, stream>>>(x, ln_in_g, ln_in_b, xn);
  bcast_templates_kernel<<<(kB * kNS * kD) / 256, 256, 0, stream>>>(t_init, tmpl);

  // ---- k|v = xn @ [Wk|Wv]   (M=32768, N=2048, K=1024) -> bf16
  gemm_bf16_wmma<<<dim3(2048 / BN, 32768 / BM, 1), 256, 0, stream>>>(
      xn, 1024, 0, Wkv_b, 2048, 0,
      nullptr, 0, 0, kv, 2048, 0, nullptr, 1.0f, 32768, 2048, 1024, 1);

  for (int it = 0; it < kIters; ++it) {
    // t = LN(templates) -> bf16
    ln_rows_kernel<<<kB * kNS, 256, 0, stream>>>(tmpl, ln_t_g, ln_t_b, tbf);
    // qT[b][d][slot] = (t @ Wq) * scale  (M=2048 fused, transposed store)
    gemm_bf16_wmma<<<dim3(1024 / BN, 2048 / BM, 1), 256, 0, stream>>>(
        tbf, 1024, 0, Wq_b, 1024, 0,
        nullptr, 0, 0, qT, 256, 262144, nullptr, kScale, 2048, 1024, 1024, 2);
    // logits[b] = k[b] @ qT[b]   (M=4096, N=256, K=1024) -> f32
    gemm_bf16_wmma<<<dim3(256 / BN, 4096 / BM, kB), 256, 0, stream>>>(
        kv, 2048, (long)4096 * 2048, qT, 256, 262144,
        logits, 256, (long)4096 * 256, nullptr, 0, 0, nullptr, 1.0f,
        4096, 256, 1024, 0);
    // softmax over N per row, +1e-8 (in place)
    softmax_rows_kernel<<<(kB * kL) / 8, 256, 0, stream>>>(logits);
    // renorm over L: column sums then divide; emit attnT bf16 + f32 output
    zero_kernel<<<8, 256, 0, stream>>>(colsum, kB * kNS);
    colsum_kernel<<<dim3(kL / 128, kB), 256, 0, stream>>>(logits, colsum);
    renorm_transpose_kernel<<<dim3(kL / 64, kNS / 64, kB), 256, 0, stream>>>(
        logits, colsum, attnT, out_attn);
    // templates += attnT[b] @ v[b]   (M=256, N=1024, K=4096)
    gemm_bf16_wmma<<<dim3(1024 / BN, 256 / BM, kB), 256, 0, stream>>>(
        attnT, 4096, (long)256 * 4096, kv + 1024, 2048, (long)4096 * 2048,
        tmpl, 1024, (long)256 * 1024, nullptr, 0, 0, nullptr, 1.0f,
        256, 1024, 4096, 4);
    // MLP: m = LN(templates); h = relu(m@W1+b1); templates += h@W2+b2
    ln_rows_kernel<<<kB * kNS, 256, 0, stream>>>(tmpl, ln_m_g, ln_m_b, tbf);
    gemm_bf16_wmma<<<dim3(512 / BN, 2048 / BM, 1), 256, 0, stream>>>(
        tbf, 1024, 0, W1_b, 512, 0,
        nullptr, 0, 0, hbf, 512, 0, b1, 1.0f, 2048, 512, 1024, 3);
    gemm_bf16_wmma<<<dim3(1024 / BN, 2048 / BM, 1), 256, 0, stream>>>(
        hbf, 512, 0, W2_b, 1024, 0,
        tmpl, 1024, 0, nullptr, 0, 0, b2, 1.0f, 2048, 1024, 512, 4);
  }

  // out_templates[b][d][n] = templates[b][n][d]
  transpose_templates_kernel<<<dim3(kD / 64, kNS / 64, kB), 256, 0, stream>>>(tmpl, out);

  (void)in_sizes; (void)n_in; (void)out_size; (void)ws_size;
}
